// CellGT_26688926777559
// MI455X (gfx1250) — matrix operand order repaired and checked by reading, not compiled, not run on hardware
//
#include <hip/hip_runtime.h>
#include <math.h>

typedef __attribute__((ext_vector_type(16))) __bf16 v16bf;
typedef __attribute__((ext_vector_type(8)))  float  v8f;

// ---------------------------------------------------------------------------
// Epilogue helper for one 16x16 C tile column (8 rows per lane).
// ---------------------------------------------------------------------------
__device__ __forceinline__ void store_ctile(float* __restrict__ C, int ldc, int M, int N,
                                            const float* __restrict__ bias,
                                            const float* __restrict__ res, int ldr, int act,
                                            v8f acc, int mbase, int n)
{
    if (n >= N) return;
#pragma unroll
    for (int v = 0; v < 8; ++v) {
        int m = mbase + v;
        if (m < M) {
            float o = acc[v];
            if (bias) o += bias[n];
            if (res)  o += res[(long long)m * ldr + n];
            if (act == 1)      o = fmaxf(o, 0.f);
            else if (act == 2) o = tanhf(o);
            C[(long long)m * ldc + n] = o;
        }
    }
}

// ---------------------------------------------------------------------------
// WMMA GEMM:  C[M,N] = act( A[M,K] @ B[K,N] + bias + res )
//   A row-major (lda), B row-major (ldb) or transposed (B[N,K], transB=1)
//   act: 0=none, 1=relu, 2=tanh.  bias/res may be null.
//   block = 128 threads (4 waves in 2x2).  Block tile 64x64; each wave owns a
//   2x2 set of 16x16 WMMA accumulators (4 v_wmma per wave per K-step; A/B
//   fragments reused across the register block).
//   Tiles staged in LDS as fp32: interior tiles via global_load_async_to_lds
//   (ASYNCcnt path), edge tiles via guarded VALU staging with zero padding.
// ---------------------------------------------------------------------------
__global__ __launch_bounds__(128) void gemm_wmma_kernel(
    const float* __restrict__ A, int lda,
    const float* __restrict__ B, int ldb, int transB,
    float* __restrict__ C, int ldc,
    int M, int N, int K,
    const float* __restrict__ bias,
    const float* __restrict__ res, int ldr,
    int act)
{
    __shared__ float As[64 * 32];   // [row 0..63][k 0..31]
    __shared__ float Bs[32 * 64];   // [k 0..31][col 0..63]
    const int tid  = threadIdx.x;
    const int wave = tid >> 5;
    const int lane = tid & 31;
    const int wr   = wave >> 1, wc = wave & 1;   // wave tile: rows 32*wr.., cols 32*wc..
    const int row0 = blockIdx.x * 64;
    const int col0 = blockIdx.y * 64;
    const int half = lane >> 4;
    const int mn   = lane & 15;

    v8f z = {0.f, 0.f, 0.f, 0.f, 0.f, 0.f, 0.f, 0.f};
    v8f acc00 = z, acc01 = z, acc10 = z, acc11 = z;   // [tm][tn]

#if defined(__AMDGCN__)
    unsigned asB = (unsigned)(size_t)&As[0];
    unsigned bsB = (unsigned)(size_t)&Bs[0];
#endif
    const bool interiorMN = (row0 + 64 <= M) && (col0 + 64 <= N);

    for (int k0 = 0; k0 < K; k0 += 32) {
        const bool interior = interiorMN && (k0 + 32 <= K);
        if (interior) {
#if defined(__AMDGCN__)
            // Async copy global->LDS, 2048 dwords per tile, 16 per lane per tile.
            // Per-lane independent global+LDS addresses also cover transB.
#pragma unroll
            for (int i = 0; i < 16; ++i) {
                int e = tid + i * 128;
                int r = e >> 5, c = e & 31;                       // A: [64][32]
                const float* ga = A + (long long)(row0 + r) * lda + (k0 + c);
                unsigned la = asB + ((unsigned)e << 2);
                asm volatile("global_load_async_to_lds_b32 %0, %1, off"
                             :: "v"(la), "v"(ga) : "memory");
                int rk = e >> 6, cn = e & 63;                     // B: [32][64]
                const float* gb = transB
                    ? (B + (long long)(col0 + cn) * ldb + (k0 + rk))
                    : (B + (long long)(k0 + rk) * ldb + (col0 + cn));
                unsigned lb = bsB + ((unsigned)e << 2);
                asm volatile("global_load_async_to_lds_b32 %0, %1, off"
                             :: "v"(lb), "v"(gb) : "memory");
            }
            asm volatile("s_wait_asynccnt 0x0" ::: "memory");
#endif
        } else {
            for (int e = tid; e < 2048; e += 128) {
                int r = e >> 5, c = e & 31;
                int gr = row0 + r, gk = k0 + c;
                As[e] = (gr < M && gk < K) ? A[(long long)gr * lda + gk] : 0.f;
                int rk = e >> 6, cn = e & 63;
                int bk = k0 + rk, bn = col0 + cn;
                float bv;
                if (transB) bv = (bn < N && bk < K) ? B[(long long)bn * ldb + bk] : 0.f;
                else        bv = (bk < K && bn < N) ? B[(long long)bk * ldb + bn] : 0.f;
                Bs[e] = bv;
            }
        }
        __syncthreads();

        // Build bf16 fragments per ISA 7.12.2 (16-bit A 16x32, B 32x16 layouts).
        v16bf a0, a1, b0, b1;
#pragma unroll
        for (int e = 0; e < 16; ++e) {
            int j = e >> 1, p = e & 1;
            int kA = ((j < 4) ? (2 * j) : (16 + 2 * (j - 4))) + 8 * half + p;
            a0[e] = (__bf16)As[(32 * wr + mn) * 32 + kA];
            a1[e] = (__bf16)As[(32 * wr + 16 + mn) * 32 + kA];
            int kB = ((e < 8) ? e : (e + 8)) + 8 * half;
            b0[e] = (__bf16)Bs[kB * 64 + 32 * wc + mn];
            b1[e] = (__bf16)Bs[kB * 64 + 32 * wc + 16 + mn];
        }
#if defined(__AMDGCN__)
        acc00 = __builtin_amdgcn_wmma_f32_16x16x32_bf16(false, a0, false, b0, (short)0, acc00, false, false);
        acc01 = __builtin_amdgcn_wmma_f32_16x16x32_bf16(false, a0, false, b1, (short)0, acc01, false, false);
        acc10 = __builtin_amdgcn_wmma_f32_16x16x32_bf16(false, a1, false, b0, (short)0, acc10, false, false);
        acc11 = __builtin_amdgcn_wmma_f32_16x16x32_bf16(false, a1, false, b1, (short)0, acc11, false, false);
#endif
        __syncthreads();
    }

    // C layout: VGPR v -> M = v + 8*half ; lane&15 -> N
    const int nA = col0 + 32 * wc + mn;
    const int nB = nA + 16;
    const int mA = row0 + 32 * wr + 8 * half;
    const int mB = mA + 16;
    store_ctile(C, ldc, M, N, bias, res, ldr, act, acc00, mA, nA);
    store_ctile(C, ldc, M, N, bias, res, ldr, act, acc01, mA, nB);
    store_ctile(C, ldc, M, N, bias, res, ldr, act, acc10, mB, nA);
    store_ctile(C, ldc, M, N, bias, res, ldr, act, acc11, mB, nB);
}

// ---------------------------------------------------------------------------
// Row-wise LayerNorm (block per row), optional fused relu.
// ---------------------------------------------------------------------------
__global__ void layernorm_kernel(const float* __restrict__ X, float* __restrict__ Y,
                                 const float* __restrict__ g, const float* __restrict__ b,
                                 int d, int relu)
{
    int row = blockIdx.x, tid = threadIdx.x;
    __shared__ float red[256];
    const float* xr = X + (long long)row * d;
    float s = 0.f;
    for (int i = tid; i < d; i += 256) s += xr[i];
    red[tid] = s; __syncthreads();
    for (int k = 128; k > 0; k >>= 1) { if (tid < k) red[tid] += red[tid + k]; __syncthreads(); }
    float m = red[0] / d; __syncthreads();
    float s2 = 0.f;
    for (int i = tid; i < d; i += 256) { float v = xr[i] - m; s2 += v * v; }
    red[tid] = s2; __syncthreads();
    for (int k = 128; k > 0; k >>= 1) { if (tid < k) red[tid] += red[tid + k]; __syncthreads(); }
    float inv = rsqrtf(red[0] / d + 1e-5f);
    float* yr = Y + (long long)row * d;
    for (int i = tid; i < d; i += 256) {
        float v = g[i] * (xr[i] - m) * inv + b[i];
        if (relu) v = fmaxf(v, 0.f);
        yr[i] = v;
    }
}

// ---------------------------------------------------------------------------
// GraphNorm (single graph): block per column, optional fused relu. In-place OK.
// ---------------------------------------------------------------------------
__global__ void graphnorm_kernel(const float* __restrict__ X, float* __restrict__ Y,
                                 const float* __restrict__ alpha, const float* __restrict__ gamma,
                                 const float* __restrict__ beta, int Nrows, int d, int relu)
{
    int c = blockIdx.x, tid = threadIdx.x;
    __shared__ float red[256];
    float s = 0.f;
    for (int i = tid; i < Nrows; i += 256) s += X[(long long)i * d + c];
    red[tid] = s; __syncthreads();
    for (int k = 128; k > 0; k >>= 1) { if (tid < k) red[tid] += red[tid + k]; __syncthreads(); }
    float m = red[0] / Nrows; __syncthreads();
    float a = alpha[c];
    float s2 = 0.f;
    for (int i = tid; i < Nrows; i += 256) { float v = X[(long long)i * d + c] - a * m; s2 += v * v; }
    red[tid] = s2; __syncthreads();
    for (int k = 128; k > 0; k >>= 1) { if (tid < k) red[tid] += red[tid + k]; __syncthreads(); }
    float inv = rsqrtf(red[0] / Nrows + 1e-5f);
    float gg = gamma[c], bb = beta[c];
    for (int i = tid; i < Nrows; i += 256) {
        float v = gg * (X[(long long)i * d + c] - a * m) * inv + bb;
        if (relu) v = fmaxf(v, 0.f);
        Y[(long long)i * d + c] = v;
    }
}

// ---------------------------------------------------------------------------
// Row-wise softmax (in-place) with scale: S[row,:] = softmax(scale*S[row,:])
// ---------------------------------------------------------------------------
__global__ void softmax_rows_kernel(float* __restrict__ S, int cols, float scale)
{
    int row = blockIdx.x, tid = threadIdx.x;
    float* sr = S + (long long)row * cols;
    __shared__ float red[256];
    float mx = -INFINITY;
    for (int i = tid; i < cols; i += 256) mx = fmaxf(mx, scale * sr[i]);
    red[tid] = mx; __syncthreads();
    for (int k = 128; k > 0; k >>= 1) { if (tid < k) red[tid] = fmaxf(red[tid], red[tid + k]); __syncthreads(); }
    mx = red[0]; __syncthreads();
    float sum = 0.f;
    for (int i = tid; i < cols; i += 256) { float e = __expf(scale * sr[i] - mx); sr[i] = e; sum += e; }
    red[tid] = sum; __syncthreads();
    for (int k = 128; k > 0; k >>= 1) { if (tid < k) red[tid] += red[tid + k]; __syncthreads(); }
    float inv = 1.f / red[0];
    for (int i = tid; i < cols; i += 256) sr[i] *= inv;
}

// ---------------------------------------------------------------------------
// GENConv softmax aggregation: out = sum_j softmax_j(t*m_j)*m_j + x,
//   m_j = relu(x[nbr_j]) + 1e-7.  One thread per (node, channel), k <= 10.
// ---------------------------------------------------------------------------
__global__ void genconv_agg_kernel(const float* __restrict__ x, const int* __restrict__ nbr,
                                   int N, int k, int d, const float* __restrict__ tptr,
                                   float* __restrict__ out)
{
    long long i = (long long)blockIdx.x * blockDim.x + threadIdx.x;
    if (i >= (long long)N * d) return;
    int node = (int)(i / d), c = (int)(i % d);
    float t = *tptr;
    float m[10];
    float mx = -INFINITY;
    for (int j = 0; j < k; ++j) {
        int s = nbr[(long long)node * k + j];
        float v = fmaxf(x[(long long)s * d + c], 0.f) + 1e-7f;
        m[j] = v;
        mx = fmaxf(mx, t * v);
    }
    float sum = 0.f, ws = 0.f;
    for (int j = 0; j < k; ++j) {
        float e = __expf(t * m[j] - mx);
        sum += e; ws += e * m[j];
    }
    out[i] = ws / sum + x[i];
}

// ---------------------------------------------------------------------------
// kNN: block per query (gridDim.x = #queries), streaming per-thread top-k
// (k <= 10) + LDS merge.  exclude_self skips i==query (knn_graph case).
// ---------------------------------------------------------------------------
__global__ void knn_kernel(const float* __restrict__ q, const float* __restrict__ base,
                           int Nb, int k, int exclude_self,
                           int* __restrict__ out_idx, float* __restrict__ out_d2)
{
    int query = blockIdx.x, tid = threadIdx.x;   // 128 threads
    float qx = q[2 * query], qy = q[2 * query + 1];
    float best[10]; int bidx[10];
    for (int j = 0; j < k; ++j) { best[j] = INFINITY; bidx[j] = 0x7fffffff; }
    for (int i = tid; i < Nb; i += 128) {
        if (exclude_self && i == query) continue;
        float dx = base[2 * i] - qx, dy = base[2 * i + 1] - qy;
        float dd = dx * dx + dy * dy;
        if (dd < best[k - 1] || (dd == best[k - 1] && i < bidx[k - 1])) {
            int j = k - 1;
            while (j > 0 && (dd < best[j - 1] || (dd == best[j - 1] && i < bidx[j - 1]))) {
                best[j] = best[j - 1]; bidx[j] = bidx[j - 1]; --j;
            }
            best[j] = dd; bidx[j] = i;
        }
    }
    __shared__ float sd[128 * 10];
    __shared__ int   si[128 * 10];
    for (int j = 0; j < k; ++j) { sd[tid * k + j] = best[j]; si[tid * k + j] = bidx[j]; }
    __syncthreads();
    if (tid == 0) {
        int total = 128 * k;
        for (int j = 0; j < k; ++j) {
            float bv = INFINITY; int bi = 0x7fffffff; int bp = 0;
            for (int t2 = 0; t2 < total; ++t2) {
                if (sd[t2] < bv || (sd[t2] == bv && si[t2] < bi)) { bv = sd[t2]; bi = si[t2]; bp = t2; }
            }
            sd[bp] = INFINITY; si[bp] = 0x7fffffff;
            out_idx[(long long)query * k + j] = bi;
            if (out_d2) out_d2[(long long)query * k + j] = bv;
        }
    }
}

// ---------------------------------------------------------------------------
// Farthest point sampling, deterministic start at 0 (single block).
// ---------------------------------------------------------------------------
__global__ void fps_kernel(const float* __restrict__ pos, int N, int n_sub,
                           int* __restrict__ out_idx, float* __restrict__ mind)
{
    __shared__ float sv[256];
    __shared__ int   si[256];
    __shared__ float bx, by;
    int tid = threadIdx.x;
    float px0 = pos[0], py0 = pos[1];
    for (int i = tid; i < N; i += 256) {
        float dx = pos[2 * i] - px0, dy = pos[2 * i + 1] - py0;
        mind[i] = dx * dx + dy * dy;
    }
    if (tid == 0) out_idx[0] = 0;
    __syncthreads();
    for (int it = 1; it < n_sub; ++it) {
        float bv = -1.f; int bi = 0x7fffffff;
        for (int i = tid; i < N; i += 256) {
            float v = mind[i];
            if (v > bv || (v == bv && i < bi)) { bv = v; bi = i; }
        }
        sv[tid] = bv; si[tid] = bi; __syncthreads();
        for (int s = 128; s > 0; s >>= 1) {
            if (tid < s) {
                if (sv[tid + s] > sv[tid] || (sv[tid + s] == sv[tid] && si[tid + s] < si[tid])) {
                    sv[tid] = sv[tid + s]; si[tid] = si[tid + s];
                }
            }
            __syncthreads();
        }
        if (tid == 0) { out_idx[it] = si[0]; bx = pos[2 * si[0]]; by = pos[2 * si[0] + 1]; }
        __syncthreads();
        float cx = bx, cy = by;
        for (int i = tid; i < N; i += 256) {
            float dx = pos[2 * i] - cx, dy = pos[2 * i + 1] - cy;
            float dd = dx * dx + dy * dy;
            if (dd < mind[i]) mind[i] = dd;
        }
        __syncthreads();
    }
}

__global__ void gather_pos_kernel(const float* __restrict__ pos, const int* __restrict__ idx,
                                  float* __restrict__ out, int n)
{
    int i = blockIdx.x * blockDim.x + threadIdx.x;
    if (i < n) { int s = idx[i]; out[2 * i] = pos[2 * s]; out[2 * i + 1] = pos[2 * s + 1]; }
}

__global__ void maxgather_kernel(const float* __restrict__ h, const int* __restrict__ nbr,
                                 int n, int k, int d, float* __restrict__ out)
{
    long long i = (long long)blockIdx.x * blockDim.x + threadIdx.x;
    if (i >= (long long)n * d) return;
    int node = (int)(i / d), c = (int)(i % d);
    float m = -INFINITY;
    for (int j = 0; j < k; ++j)
        m = fmaxf(m, h[(long long)nbr[(long long)node * k + j] * d + c]);
    out[i] = m;
}

// knn_interpolate(k=2):  X += (w0*Hs[n0] + w1*Hs[n1]) / (w0+w1), w = 1/max(d2,1e-16)
__global__ void interp_add_kernel(float* __restrict__ X, const float* __restrict__ Hs,
                                  const int* __restrict__ nbr2, const float* __restrict__ d2,
                                  int n, int d)
{
    long long i = (long long)blockIdx.x * blockDim.x + threadIdx.x;
    if (i >= (long long)n * d) return;
    int node = (int)(i / d), c = (int)(i % d);
    float w0 = 1.f / fmaxf(d2[2 * node], 1e-16f);
    float w1 = 1.f / fmaxf(d2[2 * node + 1], 1e-16f);
    float v = (w0 * Hs[(long long)nbr2[2 * node] * d + c] +
               w1 * Hs[(long long)nbr2[2 * node + 1] * d + c]) / (w0 + w1);
    X[i] += v;
}

// ---------------------------------------------------------------------------
// Fused spatial encoder: emb(96 sin/cos) -> 32 relu -> 16 tanh -> 8, concat x.
// One 32-thread block per node.
// ---------------------------------------------------------------------------
__global__ void spenc_kernel(const float* __restrict__ x, const float* __restrict__ pos,
                             const float* __restrict__ Wsp, const float* __restrict__ bsp,
                             const float* __restrict__ W1,  const float* __restrict__ b1,
                             const float* __restrict__ W2,  const float* __restrict__ b2,
                             float* __restrict__ xin, int N)
{
    int n = blockIdx.x;
    if (n >= N) return;
    int tid = threadIdx.x;   // 32
    __shared__ float emb[96];
    __shared__ float h1[32];
    __shared__ float h2[16];
    float px = pos[2 * n], py = pos[2 * n + 1];
    for (int e = tid; e < 96; e += 32) {
        int f = e / 6, s = e % 6, dir = s % 3;
        float lam = 1e-6f * powf(8.0e6f, (float)f / 15.0f);
        float dx = (dir == 0) ? 1.0f : -0.5f;
        float dy = (dir == 0) ? 0.0f : ((dir == 1) ? 0.8660254f : -0.8660254f);
        float ang = (px * dx + py * dy) / lam;
        emb[e] = (s < 3) ? sinf(ang) : cosf(ang);
    }
    __syncthreads();
    {
        float acc = bsp[tid];
        for (int i = 0; i < 96; ++i) acc += emb[i] * Wsp[i * 32 + tid];
        h1[tid] = fmaxf(acc, 0.f);
    }
    __syncthreads();
    if (tid < 16) {
        float acc = b1[tid];
        for (int i = 0; i < 32; ++i) acc += h1[i] * W1[i * 16 + tid];
        h2[tid] = tanhf(acc);
    }
    __syncthreads();
    if (tid < 24) xin[(long long)n * 32 + tid] = x[(long long)n * 24 + tid];
    if (tid < 8) {
        float acc = b2[tid];
        for (int i = 0; i < 16; ++i) acc += h2[i] * W2[i * 8 + tid];
        xin[(long long)n * 32 + 24 + tid] = acc;
    }
}

// ===========================================================================
// Host orchestration
// ===========================================================================
struct LinP { const float* W; const float* b; };
struct LNP  { const float* g; const float* b; };
struct GNP  { const float* alpha; const float* gamma; const float* beta; };
struct GenP { const float* t; LinP lin1; LNP ln; LinP lin2; };
struct TFP  { LinP Wq, Wk, Wv, Wo, ff1, ff2; LNP ln1, ln2; };

extern "C" void kernel_launch(void* const* d_in, const int* in_sizes, int n_in,
                              void* d_out, int out_size, void* d_ws, size_t ws_size,
                              hipStream_t stream)
{
    (void)n_in; (void)out_size; (void)ws_size;
    const float* x   = (const float*)d_in[0];
    const float* pos = (const float*)d_in[1];
    // d_in[2] = batch (all zeros, unused)

    const int N0 = in_sizes[0] / 24;
    const int n1 = (N0 + 5) / 6;
    const int n2 = (n1 + 5) / 6;
    const int n3 = (n2 + 5) / 6;
    const int D0 = 128, D1 = 256, D2 = 512, D3 = 1024;
    const int K0 = 10, K1 = 6, K2 = 3;   // LIST_K[0..2]; LIST_K[3] unused (tf level)

    // ---- parameter walker (setup_inputs insertion order) ----
    int cur = 3;
    auto nf     = [&]() { return (const float*)d_in[cur++]; };
    auto getLin = [&]() { LinP l; l.W = nf(); l.b = nf(); return l; };
    auto getLN  = [&]() { LNP l; l.g = nf(); l.b = nf(); return l; };
    auto getGN  = [&]() { GNP g; g.alpha = nf(); g.gamma = nf(); g.beta = nf(); return g; };
    auto getGen = [&]() { GenP g; g.t = nf(); g.lin1 = getLin(); g.ln = getLN(); g.lin2 = getLin(); return g; };
    auto getTF  = [&]() { TFP t; t.Wq = getLin(); t.Wk = getLin(); t.Wv = getLin(); t.Wo = getLin();
                          t.ff1 = getLin(); t.ff2 = getLin(); t.ln1 = getLN(); t.ln2 = getLN(); return t; };

    LinP spenc = getLin(), dec1 = getLin(), dec2 = getLin();
    LinP mlpin_lin = getLin(); GNP mlpin_gn = getGN();
    GenP in_conv0 = getGen();
    GenP in_conv1_conv = getGen(); LNP in_conv1_ln = getLN();
    LinP td_lin[3]; GNP td_gn[3];
    for (int i = 0; i < 3; ++i) { td_lin[i] = getLin(); td_gn[i] = getGN(); }
    GenP down_conv[2]; LNP down_ln0[2], down_ln1[2];
    for (int i = 0; i < 2; ++i) { down_conv[i] = getGen(); down_ln0[i] = getLN(); down_ln1[i] = getLN(); }
    TFP tf_down[2]; for (int i = 0; i < 2; ++i) tf_down[i] = getTF();
    LinP summit = getLin();
    TFP tf_summit[4]; for (int i = 0; i < 4; ++i) tf_summit[i] = getTF();
    LinP tu_sub_lin[3]; GNP tu_sub_gn[3]; LinP tu_mlp_lin[3]; GNP tu_mlp_gn[3];
    for (int i = 0; i < 3; ++i) { tu_sub_lin[i] = getLin(); tu_sub_gn[i] = getGN();
                                  tu_mlp_lin[i] = getLin(); tu_mlp_gn[i] = getGN(); }
    TFP tf_up[2]; for (int i = 0; i < 2; ++i) tf_up[i] = getTF();
    GenP up_conv[2]; LNP up_ln0[2], up_ln1[2];
    for (int i = 0; i < 2; ++i) { up_conv[i] = getGen(); up_ln0[i] = getLN(); up_ln1[i] = getLN(); }
    GenP oc_conv[4]; LNP oc_ln[4];
    for (int i = 0; i < 4; ++i) { oc_conv[i] = getGen(); oc_ln[i] = getLN(); }
    LinP mlp_out = getLin();

    // ---- workspace arena ----
    char* wsb = (char*)d_ws; size_t off = 0;
    auto af = [&](size_t n) { float* p = (float*)(wsb + off); off += ((n * 4 + 255) / 256) * 256; return p; };
    auto ai = [&](size_t n) { int*   p = (int*)(wsb + off);   off += ((n * 4 + 255) / 256) * 256; return p; };

    float* xin = af((size_t)N0 * 32);
    float* x0  = af((size_t)N0 * D0);   // level-0 skip (out_x[0])
    float* x1  = af((size_t)n1 * D1);   // level-1 skip
    float* x2  = af((size_t)n2 * D2);   // level-2 skip
    float* x3  = af((size_t)n3 * D3);
    float* y3  = af((size_t)n3 * D3);
    float* up2 = af((size_t)n2 * D2);
    float* up1 = af((size_t)n1 * D1);
    float* up0 = af((size_t)N0 * D0);
    float* p1  = af((size_t)n1 * 2);
    float* p2  = af((size_t)n2 * 2);
    float* p3  = af((size_t)n3 * 2);
    int* nbr0   = ai((size_t)N0 * K0);
    int* nbr1   = ai((size_t)n1 * K1);
    int* nbr2g  = ai((size_t)n2 * K2);
    int* tdnbr  = ai((size_t)n1 * 10);
    int* fpsidx = ai((size_t)n1);
    float* mind = af((size_t)N0);
    int*   nnI  = ai((size_t)N0 * 2);
    float* nnD  = af((size_t)N0 * 2);
    size_t hsN = (size_t)n1 * D0;
    if ((size_t)n2 * D1 > hsN) hsN = (size_t)n2 * D1;
    if ((size_t)n3 * D2 > hsN) hsN = (size_t)n3 * D2;
    float* Hs   = af(hsN);
    float* tmpA = af((size_t)N0 * D0);
    float* tmpB = af((size_t)N0 * D0);
    float* tmpC = af((size_t)N0 * 2 * D0);
    size_t tfMd = (size_t)n2 * D3;
    float* Qb = af(tfMd); float* Kb = af(tfMd); float* Vb = af(tfMd);
    float* Ob = af(tfMd); float* Tb = af(tfMd);
    float* Fb = af((size_t)n2 * 2048);
    float* Sb = af((size_t)n2 * n2);

    // ---- launch helpers ----
    auto gemm = [&](const float* A, int lda, const float* B, int ldb, int tB,
                    float* C, int ldc, int M, int N, int K,
                    const float* bias, const float* res, int ldr, int act) {
        dim3 g((M + 63) / 64, (N + 63) / 64);
        gemm_wmma_kernel<<<g, 128, 0, stream>>>(A, lda, B, ldb, tB, C, ldc, M, N, K, bias, res, ldr, act);
    };
    auto layernorm = [&](const float* X, float* Y, LNP p, int M, int d, int relu) {
        layernorm_kernel<<<M, 256, 0, stream>>>(X, Y, p.g, p.b, d, relu);
    };
    auto graphnorm = [&](float* X, GNP p, int M, int d, int relu) {
        graphnorm_kernel<<<d, 256, 0, stream>>>(X, X, p.alpha, p.gamma, p.beta, M, d, relu);
    };
    auto mlp_gn = [&](const float* in, int M, int din, LinP l, GNP g, float* out, int dout) {
        gemm(in, din, l.W, dout, 0, out, dout, M, dout, din, l.b, nullptr, 0, 0);
        graphnorm(out, g, M, dout, 1);
    };
    auto genagg = [&](const float* xb, const int* nbr, int N, int k, int d, const float* t, float* out) {
        long long tot = (long long)N * d;
        genconv_agg_kernel<<<(int)((tot + 255) / 256), 256, 0, stream>>>(xb, nbr, N, k, d, t, out);
    };
    auto deepgcn = [&](float* xb, const int* nbr, int N, int k, int d, const GenP& g, LNP ln, int outer_relu) {
        layernorm(xb, tmpA, ln, N, d, 1);                                  // h = relu(LN(x))
        genagg(tmpA, nbr, N, k, d, g.t, tmpB);                             // softmax aggr + root
        gemm(tmpB, d, g.lin1.W, 2 * d, 0, tmpC, 2 * d, N, 2 * d, d, g.lin1.b, nullptr, 0, 0);
        layernorm(tmpC, tmpC, g.ln, N, 2 * d, 1);
        gemm(tmpC, 2 * d, g.lin2.W, d, 0, xb, d, N, d, 2 * d, g.lin2.b, xb, d, outer_relu);
    };
    auto tflayer = [&](float* xb, int M, int d, const TFP& p) {
        int dh = d / 2;
        float scale = 1.0f / sqrtf((float)dh);
        gemm(xb, d, p.Wq.W, d, 0, Qb, d, M, d, d, p.Wq.b, nullptr, 0, 0);
        gemm(xb, d, p.Wk.W, d, 0, Kb, d, M, d, d, p.Wk.b, nullptr, 0, 0);
        gemm(xb, d, p.Wv.W, d, 0, Vb, d, M, d, d, p.Wv.b, nullptr, 0, 0);
        for (int h = 0; h < 2; ++h) {
            gemm(Qb + h * dh, d, Kb + h * dh, d, 1, Sb, M, M, M, dh, nullptr, nullptr, 0, 0);
            softmax_rows_kernel<<<M, 256, 0, stream>>>(Sb, M, scale);
            gemm(Sb, M, Vb + h * dh, d, 0, Ob + h * dh, d, M, dh, M, nullptr, nullptr, 0, 0);
        }
        gemm(Ob, d, p.Wo.W, d, 0, Tb, d, M, d, d, p.Wo.b, xb, d, 0);      // x + attn
        layernorm(Tb, xb, p.ln1, M, d, 0);
        gemm(xb, d, p.ff1.W, 2048, 0, Fb, 2048, M, 2048, d, p.ff1.b, nullptr, 0, 1);
        gemm(Fb, 2048, p.ff2.W, d, 0, Tb, d, M, d, 2048, p.ff2.b, xb, d, 0);
        layernorm(Tb, xb, p.ln2, M, d, 0);
    };
    auto ewgrid = [&](long long tot) { return (int)((tot + 255) / 256); };

    // ================= forward pass =================
    spenc_kernel<<<N0, 32, 0, stream>>>(x, pos, spenc.W, spenc.b, dec1.W, dec1.b, dec2.W, dec2.b, xin, N0);
    mlp_gn(xin, N0, 32, mlpin_lin, mlpin_gn, x0, D0);
    knn_kernel<<<N0, 128, 0, stream>>>(pos, pos, N0, K0, 1, nbr0, nullptr);

    // in_conv0: raw GENConv
    genagg(x0, nbr0, N0, K0, D0, in_conv0.t, tmpB);
    gemm(tmpB, D0, in_conv0.lin1.W, 2 * D0, 0, tmpC, 2 * D0, N0, 2 * D0, D0, in_conv0.lin1.b, nullptr, 0, 0);
    layernorm(tmpC, tmpC, in_conv0.ln, N0, 2 * D0, 1);
    gemm(tmpC, 2 * D0, in_conv0.lin2.W, D0, 0, x0, D0, N0, D0, 2 * D0, in_conv0.lin2.b, nullptr, 0, 0);
    // in_conv1: DeepGCN res+
    deepgcn(x0, nbr0, N0, K0, D0, in_conv1_conv, in_conv1_ln, 0);

    // ---- down 0 -> 1
    mlp_gn(x0, N0, D0, td_lin[0], td_gn[0], tmpC, D1);
    fps_kernel<<<1, 256, 0, stream>>>(pos, N0, n1, fpsidx, mind);
    gather_pos_kernel<<<(n1 + 255) / 256, 256, 0, stream>>>(pos, fpsidx, p1, n1);
    knn_kernel<<<n1, 128, 0, stream>>>(p1, pos, N0, K0, 0, tdnbr, nullptr);
    maxgather_kernel<<<ewgrid((long long)n1 * D1), 256, 0, stream>>>(tmpC, tdnbr, n1, K0, D1, x1);
    knn_kernel<<<n1, 128, 0, stream>>>(p1, p1, n1, K1, 1, nbr1, nullptr);
    deepgcn(x1, nbr1, n1, K1, D1, down_conv[0], down_ln0[0], 1);
    deepgcn(x1, nbr1, n1, K1, D1, down_conv[0], down_ln1[0], 1);

    // ---- down 1 -> 2
    mlp_gn(x1, n1, D1, td_lin[1], td_gn[1], tmpC, D2);
    fps_kernel<<<1, 256, 0, stream>>>(p1, n1, n2, fpsidx, mind);
    gather_pos_kernel<<<(n2 + 255) / 256, 256, 0, stream>>>(p1, fpsidx, p2, n2);
    knn_kernel<<<n2, 128, 0, stream>>>(p2, p1, n1, K1, 0, tdnbr, nullptr);
    maxgather_kernel<<<ewgrid((long long)n2 * D2), 256, 0, stream>>>(tmpC, tdnbr, n2, K1, D2, x2);
    knn_kernel<<<n2, 128, 0, stream>>>(p2, p2, n2, K2, 1, nbr2g, nullptr);
    deepgcn(x2, nbr2g, n2, K2, D2, down_conv[1], down_ln0[1], 1);
    deepgcn(x2, nbr2g, n2, K2, D2, down_conv[1], down_ln1[1], 1);

    // ---- down 2 -> 3 (transformer level)
    mlp_gn(x2, n2, D2, td_lin[2], td_gn[2], tmpC, D3);
    fps_kernel<<<1, 256, 0, stream>>>(p2, n2, n3, fpsidx, mind);
    gather_pos_kernel<<<(n3 + 255) / 256, 256, 0, stream>>>(p2, fpsidx, p3, n3);
    knn_kernel<<<n3, 128, 0, stream>>>(p3, p2, n2, K2, 0, tdnbr, nullptr);
    maxgather_kernel<<<ewgrid((long long)n3 * D3), 256, 0, stream>>>(tmpC, tdnbr, n3, K2, D3, x3);
    tflayer(x3, n3, D3, tf_down[0]);
    tflayer(x3, n3, D3, tf_down[1]);

    // ---- summit
    gemm(x3, D3, summit.W, D3, 0, y3, D3, n3, D3, D3, summit.b, nullptr, 0, 1);
    for (int i = 0; i < 4; ++i) tflayer(y3, n3, D3, tf_summit[i]);

    // ---- up 3 -> 2 (transformer branch)
    mlp_gn(y3, n3, D3, tu_sub_lin[2], tu_sub_gn[2], Hs, D2);
    knn_kernel<<<n2, 128, 0, stream>>>(p2, p3, n3, 2, 0, nnI, nnD);
    mlp_gn(x2, n2, D2, tu_mlp_lin[2], tu_mlp_gn[2], up2, D2);
    interp_add_kernel<<<ewgrid((long long)n2 * D2), 256, 0, stream>>>(up2, Hs, nnI, nnD, n2, D2);
    tflayer(up2, n2, D2, tf_up[0]);
    tflayer(up2, n2, D2, tf_up[1]);

    // ---- up 2 -> 1 (conv branch)
    mlp_gn(up2, n2, D2, tu_sub_lin[1], tu_sub_gn[1], Hs, D1);
    knn_kernel<<<n1, 128, 0, stream>>>(p1, p2, n2, 2, 0, nnI, nnD);
    mlp_gn(x1, n1, D1, tu_mlp_lin[1], tu_mlp_gn[1], up1, D1);
    interp_add_kernel<<<ewgrid((long long)n1 * D1), 256, 0, stream>>>(up1, Hs, nnI, nnD, n1, D1);
    deepgcn(up1, nbr1, n1, K1, D1, up_conv[1], up_ln0[1], 1);
    deepgcn(up1, nbr1, n1, K1, D1, up_conv[1], up_ln1[1], 1);

    // ---- up 1 -> 0
    mlp_gn(up1, n1, D1, tu_sub_lin[0], tu_sub_gn[0], Hs, D0);
    knn_kernel<<<N0, 128, 0, stream>>>(pos, p1, n1, 2, 0, nnI, nnD);
    mlp_gn(x0, N0, D0, tu_mlp_lin[0], tu_mlp_gn[0], up0, D0);
    interp_add_kernel<<<ewgrid((long long)N0 * D0), 256, 0, stream>>>(up0, Hs, nnI, nnD, N0, D0);
    deepgcn(up0, nbr0, N0, K0, D0, up_conv[0], up_ln0[0], 1);
    deepgcn(up0, nbr0, N0, K0, D0, up_conv[0], up_ln1[0], 0);

    // ---- out convs + head
    for (int i = 0; i < 4; ++i)
        deepgcn(up0, nbr0, N0, K0, D0, oc_conv[i], oc_ln[i], 0);
    gemm(up0, D0, mlp_out.W, 8, 0, (float*)d_out, 8, N0, 8, D0, mlp_out.b, nullptr, 0, 0);
}
